// Shortcut_BinConv_BN_88991722373188
// MI455X (gfx1250) — 8-bit matrix operand order repaired and checked by reading, compiled, not run on hardware
//
#include <hip/hip_runtime.h>
#include <stdint.h>

typedef __attribute__((ext_vector_type(16))) int   v16i;
typedef __attribute__((ext_vector_type(8)))  float v8f;

#define C_CH   128
#define HW     56
#define PW     58
#define BATCH  32
#define M_TOTAL (BATCH*HW*HW)
#define BN_EPS 1e-3f

#define FP8_P1 0x38u
#define FP8_M1 0xB8u

#define XB_OFF    0
#define XB_BYTES  (BATCH*PW*PW*C_CH)
#define WF_OFF    (XB_OFF + XB_BYTES)
#define WF_BYTES  (8*9*16*32*4)
#define R_OFF     (WF_OFF + WF_BYTES)
#define R_BYTES   ((size_t)M_TOTAL*C_CH*2)
#define SUM_OFF   (R_OFF + R_BYTES)
#define SCALE_OFF (SUM_OFF + 256*8)
#define BIAS_OFF  (SCALE_OFF + 128*4)

__global__ __launch_bounds__(256) void binarize_pad_kernel(
    const float* __restrict__ x, uint32_t* __restrict__ xb4,
    unsigned long long* __restrict__ sums) {
  if (blockIdx.x == 0 && threadIdx.x < 256) sums[threadIdx.x] = 0ull;
  int idx = blockIdx.x * 256 + threadIdx.x;
  const int WORDS = BATCH * PW * PW * (C_CH / 4);
  if (idx >= WORDS) return;
  int c4 = idx & 31;
  int t  = idx >> 5;
  int px = t % PW; t /= PW;
  int py = t % PW; int b = t / PW;
  uint32_t w = 0u;
  if (py >= 1 && py <= HW && px >= 1 && px <= HW) {
    const float4 f = *(const float4*)(x + ((size_t)((b*HW + (py-1))*HW + (px-1)))*C_CH + c4*4);
    uint32_t b0 = (f.x >= 0.f) ? FP8_P1 : FP8_M1;
    uint32_t b1 = (f.y >= 0.f) ? FP8_P1 : FP8_M1;
    uint32_t b2 = (f.z >= 0.f) ? FP8_P1 : FP8_M1;
    uint32_t b3 = (f.w >= 0.f) ? FP8_P1 : FP8_M1;
    w = b0 | (b1 << 8) | (b2 << 16) | (b3 << 24);
  }
  xb4[idx] = w;
}

__global__ __launch_bounds__(256) void wpack_kernel(
    const float* __restrict__ W, uint32_t* __restrict__ wf) {
  int idx = blockIdx.x * 256 + threadIdx.x;
  if (idx >= 8*9*16*32) return;
  int nw   = idx / 4608;
  int r    = idx % 4608;
  int s    = r / 512;
  int i    = (r >> 5) & 15;
  int lane = r & 31;
  int n    = nw * 16 + (lane & 15);
  int half = lane >> 4;
  int kk   = ((i >> 2) << 5) + (half << 4) + ((i & 3) << 2);
  int k    = s * 128 + kk;
  uint32_t w = 0u;
#pragma unroll
  for (int j = 0; j < 4; ++j) {
    uint32_t sb = (W[(size_t)(k + j) * C_CH + n] >= 0.f) ? FP8_P1 : FP8_M1;
    w |= sb << (8 * j);
  }
  wf[idx] = w;
}

__global__ __launch_bounds__(256) void bingemm_kernel(
    const int8_t* __restrict__ xb, const uint32_t* __restrict__ wf,
    short* __restrict__ rws, unsigned long long* __restrict__ sums) {
  __shared__ __align__(16) uint8_t tileA[3 * PW * C_CH];

  const int wg = blockIdx.x;
  const int b = wg / HW, y = wg % HW;

  {
    const uint4* src = (const uint4*)(xb + (size_t)(b * PW + y) * PW * C_CH);
    uint4* dst = (uint4*)tileA;
    for (int t = threadIdx.x; t < (3 * PW * C_CH) / 16; t += 256) dst[t] = src[t];
  }
  __syncthreads();

  const int lane = threadIdx.x & 31;
  const int wave = threadIdx.x >> 5;
  const int lm = lane & 15, half = lane >> 4;
  const int nbase = wave * 16;

  int koff[16];
#pragma unroll
  for (int i = 0; i < 16; ++i) {
    int il = i & 7;
    koff[i] = ((i >> 2) << 5) + (half << 4) + ((i & 3) << 2);
  }

  int xbase[4];
#pragma unroll
  for (int t = 0; t < 4; ++t) { int xx = t * 16 + lm; if (xx > 55) xx = 55; xbase[t] = xx * C_CH; }

  v8f acc[4];
#pragma unroll
  for (int t = 0; t < 4; ++t)
#pragma unroll
    for (int v = 0; v < 8; ++v) acc[t][v] = 0.f;

  const uint32_t* wfw = wf + wave * 4608;

#pragma unroll 1
  for (int s = 0; s < 9; ++s) {
    v16i bfrag;
#pragma unroll
    for (int i = 0; i < 16; ++i) bfrag[i] = (int)wfw[s * 512 + i * 32 + lane];

    int addr[16];
#pragma unroll
    for (int i = 0; i < 16; ++i) {
      int k = s * 128 + koff[i];
      int dy = k / 384;
      addr[i] = dy * (PW * C_CH) + (k - dy * 384);
    }
#pragma unroll
    for (int t = 0; t < 4; ++t) {
      v16i a;
#pragma unroll
      for (int i = 0; i < 16; ++i) a[i] = *(const int*)(tileA + addr[i] + xbase[t]);
      acc[t] = __builtin_amdgcn_wmma_f32_16x16x128_fp8_fp8(
          a, bfrag,  (short)0, acc[t],
           false,  false);
    }
  }

  int sum = 0, sq = 0;
  const int n = nbase + lm;
  short* rp = rws + (size_t)((b * HW + y) * HW) * C_CH;
#pragma unroll
  for (int t = 0; t < 4; ++t) {
    if (!(t == 3 && half == 1)) {
#pragma unroll
      for (int v = 0; v < 8; ++v) {
        int xx = t * 16 + half * 8 + v;
        float fr = acc[t][v];
        int ri = fr > 0.f ? (int)fr : 0;
        sum += ri; sq += ri * ri;
        rp[(size_t)xx * C_CH + n] = (short)ri;
      }
    }
  }
  sum += __shfl_xor(sum, 16, 32);
  sq  += __shfl_xor(sq,  16, 32);
  if (half == 0) {
    atomicAdd(&sums[n],       (unsigned long long)sum);
    atomicAdd(&sums[128 + n], (unsigned long long)sq);
  }
}

__global__ void bnstat_kernel(const unsigned long long* __restrict__ sums,
                              const float* __restrict__ gamma, const float* __restrict__ beta,
                              float* __restrict__ scale, float* __restrict__ bias) {
  int c = threadIdx.x;
  if (c < C_CH) {
    const double M = (double)M_TOTAL;
    double s = (double)sums[c], q = (double)sums[128 + c];
    double mean = s / M;
    double var = q / M - mean * mean;
    float sc = gamma[c] * rsqrtf((float)var + BN_EPS);
    scale[c] = sc;
    bias[c] = beta[c] - (float)mean * sc;
  }
}

__global__ __launch_bounds__(256) void apply_kernel(
    const short* __restrict__ r, const float* __restrict__ x,
    const float* __restrict__ scale, const float* __restrict__ bias,
    float* __restrict__ out) {
  size_t e = ((size_t)blockIdx.x * 256 + threadIdx.x) * 4;
  if (e >= (size_t)M_TOTAL * C_CH) return;
  int c0 = (int)(e & (C_CH - 1));
  short4 rv = *(const short4*)(r + e);
  float4 xv = *(const float4*)(x + e);
  float4 sc = *(const float4*)(scale + c0);
  float4 bi = *(const float4*)(bias + c0);
  float4 o;
  o.x = sc.x * (float)rv.x + bi.x + xv.x;
  o.y = sc.y * (float)rv.y + bi.y + xv.y;
  o.z = sc.z * (float)rv.z + bi.z + xv.z;
  o.w = sc.w * (float)rv.w + bi.w + xv.w;
  *(float4*)(out + e) = o;
}

extern "C" void kernel_launch(void* const* d_in, const int* in_sizes, int n_in,
                              void* d_out, int out_size, void* d_ws, size_t ws_size,
                              hipStream_t stream) {
  const float* x     = (const float*)d_in[0];
  const float* W     = (const float*)d_in[1];
  const float* gamma = (const float*)d_in[2];
  const float* beta  = (const float*)d_in[3];
  float* out = (float*)d_out;

  char* ws = (char*)d_ws;
  int8_t*             xb    = (int8_t*)(ws + XB_OFF);
  uint32_t*           wf    = (uint32_t*)(ws + WF_OFF);
  short*              r     = (short*)(ws + R_OFF);
  unsigned long long* sums  = (unsigned long long*)(ws + SUM_OFF);
  float*              scale = (float*)(ws + SCALE_OFF);
  float*              bias  = (float*)(ws + BIAS_OFF);

  const int padWords = BATCH * PW * PW * (C_CH / 4);
  binarize_pad_kernel<<<(padWords + 255) / 256, 256, 0, stream>>>(x, (uint32_t*)xb, sums);
  wpack_kernel<<<(8 * 9 * 16 * 32 + 255) / 256, 256, 0, stream>>>(W, wf);
  bingemm_kernel<<<BATCH * HW, 256, 0, stream>>>(xb, wf, r, sums);
  bnstat_kernel<<<1, 128, 0, stream>>>(sums, gamma, beta, scale, bias);
  apply_kernel<<<(int)(((size_t)M_TOTAL * C_CH / 4 + 255) / 256), 256, 0, stream>>>(r, x, scale, bias, out);
}
